// TransformerNet_46540265619754
// MI455X (gfx1250) — matrix operand order repaired and checked by reading, not compiled, not run on hardware
//
#include <hip/hip_runtime.h>
#include <hip/hip_bf16.h>
#include <math.h>

typedef __attribute__((ext_vector_type(16))) __bf16 v16bf;
typedef __attribute__((ext_vector_type(8)))  float  v8f;

#define NN 10000
#define NE 160000
#define NG 16

// ---------------- elementwise helpers ----------------

__global__ void f32_to_bf16_k(const float* __restrict__ x, __bf16* __restrict__ y, int n) {
  int i = blockIdx.x * blockDim.x + threadIdx.x;
  if (i < n) y[i] = (__bf16)x[i];
}

// Pack 4 weight matrices [fin x fout] (row-major) into transposed bf16 rows:
// wtp[n][k], n in [0,4*fout): block q|k|v|s. bp[n] = matching bias.
__global__ void pack_weights_k(const float* __restrict__ Wq, const float* __restrict__ Wk,
                               const float* __restrict__ Wv, const float* __restrict__ Ws,
                               const float* __restrict__ bq, const float* __restrict__ bk,
                               const float* __restrict__ bv, const float* __restrict__ bs,
                               __bf16* __restrict__ wtp, float* __restrict__ bp,
                               int fin, int fout) {
  int idx = blockIdx.x * blockDim.x + threadIdx.x;
  int total = 4 * fout * fin;
  if (idx >= total) return;
  int n = idx / fin, k = idx % fin;
  int blk = n / fout, col = n % fout;
  const float* W = (blk == 0) ? Wq : (blk == 1) ? Wk : (blk == 2) ? Wv : Ws;
  wtp[(size_t)n * fin + k] = (__bf16)W[(size_t)k * fout + col];
  if (k == 0) {
    const float* b = (blk == 0) ? bq : (blk == 1) ? bk : (blk == 2) ? bv : bs;
    bp[n] = b[col];
  }
}

// ---------------- WMMA GEMM: C[M x Np] = A[M x K](bf16) * Bt[Np x K]^T + bias -------------
// One wave -> 32x64 output strip: 2 M-tiles x 4 N-tiles, double-buffered K pipeline.
struct Frag { union { int4 i4[2]; v16bf v; }; };

__global__ __launch_bounds__(256)
void wmma_gemm_bias_k(const __bf16* __restrict__ A, const __bf16* __restrict__ Bt,
                      const float* __restrict__ bias, float* __restrict__ C,
                      int M, int K, int Np) {
  const int wave = threadIdx.x >> 5;
  const int lane = threadIdx.x & 31;
  const int hf   = lane >> 4;     // half-wave select
  const int l16  = lane & 15;
  const int mtile = blockIdx.x * 8 + wave;     // 32-row tile
  const int mBase = mtile * 32;
  if (mBase >= M) return;         // wave-uniform: EXEC stays all-ones for WMMA
  const int nBase = blockIdx.y * 64;

  v8f c0[4], c1[4];
#pragma unroll
  for (int t = 0; t < 4; ++t) {
    float bv = bias[nBase + t * 16 + l16];   // bias is per output column (N)
#pragma unroll
    for (int r = 0; r < 8; ++r) { c0[t][r] = bv; c1[t][r] = bv; }
  }

  int rowA1 = mBase + 16 + l16; if (rowA1 > M - 1) rowA1 = M - 1;  // clamp tail (uniform EXEC)
  const __bf16* arow0 = A + (size_t)(mBase + l16) * K;
  const __bf16* arow1 = A + (size_t)rowA1 * K;
  const __bf16* brow[4];
#pragma unroll
  for (int t = 0; t < 4; ++t) brow[t] = Bt + (size_t)(nBase + t * 16 + l16) * K;

  // A 16x32 fragment (ISA 7.12.2): lanes0-15 K 0-7 & 16-23; lanes16-31 K 8-15 & 24-31
  // B 32x16 fragment: lane = N; lanes0-15 K 0-7 & 16-23; lanes16-31 K 8-15 & 24-31
  auto loadset = [&](Frag& fa0, Frag& fa1, Frag* fb, int kk) {
    fa0.i4[0] = *(const int4*)(arow0 + kk + hf * 8);
    fa0.i4[1] = *(const int4*)(arow0 + kk + 16 + hf * 8);
    fa1.i4[0] = *(const int4*)(arow1 + kk + hf * 8);
    fa1.i4[1] = *(const int4*)(arow1 + kk + 16 + hf * 8);
#pragma unroll
    for (int t = 0; t < 4; ++t) {
      const __bf16* p = brow[t] + kk + hf * 8;
      fb[t].i4[0] = ((const int4*)p)[0];
      fb[t].i4[1] = ((const int4*)(p + 16))[0];
    }
  };
  auto compute = [&](Frag& fa0, Frag& fa1, Frag* fb) {
#pragma unroll
    for (int t = 0; t < 4; ++t) {
      c0[t] = __builtin_amdgcn_wmma_f32_16x16x32_bf16(false, fa0.v, false, fb[t].v,
                                                      (short)0, c0[t], false, false);
      c1[t] = __builtin_amdgcn_wmma_f32_16x16x32_bf16(false, fa1.v, false, fb[t].v,
                                                      (short)0, c1[t], false, false);
    }
  };

  Frag pa0, pa1, pb[4];      // pipeline buffer A (even steps)
  Frag qa0, qa1, qb[4];      // pipeline buffer B (odd steps)
  loadset(pa0, pa1, pb, 0);
  for (int kk = 0; kk < K; kk += 64) {        // all K are multiples of 64
    loadset(qa0, qa1, qb, kk + 32);           // next step's loads in flight...
    if (kk + 128 < K) {                       // WGP-scope stream-ahead prefetch
      __builtin_prefetch(arow0 + kk + 128, 0, 3);   // global_prefetch_b8, fills WGP$
      __builtin_prefetch(arow1 + kk + 128, 0, 3);
    }
    compute(pa0, pa1, pb);                    // ...while these 8 WMMAs execute
    if (kk + 64 < K) loadset(pa0, pa1, pb, kk + 64);
    compute(qa0, qa1, qb);
  }

#pragma unroll
  for (int t = 0; t < 4; ++t) {
    int col = nBase + t * 16 + l16;
#pragma unroll
    for (int r = 0; r < 8; ++r)   // C layout: VGPR r -> row r (lanes0-15) / 8+r (lanes16-31)
      C[(size_t)(mBase + hf * 8 + r) * Np + col] = c0[t][r];
  }
  if (mBase + 16 < M) {           // second M-tile valid only off the tail
#pragma unroll
    for (int t = 0; t < 4; ++t) {
      int col = nBase + t * 16 + l16;
#pragma unroll
      for (int r = 0; r < 8; ++r)
        C[(size_t)(mBase + 16 + hf * 8 + r) * Np + col] = c1[t][r];
    }
  }
}

// ---------------- edge attention (segment softmax over dst) ----------------

__global__ void init_node_k(float* __restrict__ m, float* __restrict__ s, int n) {
  int i = blockIdx.x * blockDim.x + threadIdx.x;
  if (i < n) { m[i] = -INFINITY; s[i] = 0.f; }
}

// wave per edge: logits[e] = dot(q[dst], k[src]) * scale ; atomicMax into segment max
__global__ void edge_logits_k(const float* __restrict__ qkvs, const int* __restrict__ src,
                              const int* __restrict__ dst, float* __restrict__ logits,
                              float* __restrict__ m, int E, int fout, int ldc, float scale) {
  int e = (blockIdx.x * blockDim.x + threadIdx.x) >> 5;
  int lane = threadIdx.x & 31;
  if (e >= E) return;
  int de = dst[e], se = src[e];
  const float* q = qkvs + (size_t)de * ldc;          // q at col 0
  const float* k = qkvs + (size_t)se * ldc + fout;   // k at col fout
  float sum = 0.f;
  for (int c = lane * 4; c < fout; c += 128) {       // b128 gathers
    float4 a = *(const float4*)(q + c);
    float4 b = *(const float4*)(k + c);
    sum += a.x * b.x + a.y * b.y + a.z * b.z + a.w * b.w;
  }
  for (int off = 16; off; off >>= 1) sum += __shfl_down(sum, off, 32);
  if (lane == 0) {
    float v = sum * scale;
    logits[e] = v;
    atomicMax(&m[de], v);                            // global_atomic_max_num_f32
  }
}

__global__ void edge_exp_k(const float* __restrict__ logits, const float* __restrict__ m,
                           float* __restrict__ s, float* __restrict__ ebuf,
                           const int* __restrict__ dst, int E) {
  int e = blockIdx.x * blockDim.x + threadIdx.x;
  if (e >= E) return;
  int de = dst[e];
  float v = expf(logits[e] - m[de]);
  ebuf[e] = v;
  atomicAdd(&s[de], v);
}

// agg[n][c] = skip projection S (already has bias); float4 copy
__global__ void init_agg_skip_k(const float* __restrict__ qkvs, float* __restrict__ agg,
                                int fout, int ldc, int total4) {
  int i = blockIdx.x * blockDim.x + threadIdx.x;
  if (i >= total4) return;
  int f4 = fout >> 2;
  int n = i / f4, c = (i % f4) << 2;
  *(float4*)(agg + (size_t)n * fout + c) =
      *(const float4*)(qkvs + (size_t)n * ldc + 3 * fout + c);
}

// wave per edge: agg[dst] += (e/s[dst]) * v[src]
__global__ void edge_scatter_k(const float* __restrict__ qkvs, const float* __restrict__ ebuf,
                               const float* __restrict__ s, const int* __restrict__ src,
                               const int* __restrict__ dst, float* __restrict__ agg,
                               int E, int fout, int ldc) {
  int e = (blockIdx.x * blockDim.x + threadIdx.x) >> 5;
  int lane = threadIdx.x & 31;
  if (e >= E) return;
  int de = dst[e], se = src[e];
  float alpha = ebuf[e] / s[de];
  const float* v = qkvs + (size_t)se * ldc + 2 * fout;
  float* ag = agg + (size_t)de * fout;
  for (int c = lane * 4; c < fout; c += 128) {
    float4 vv = *(const float4*)(v + c);
    atomicAdd(&ag[c + 0], alpha * vv.x);
    atomicAdd(&ag[c + 1], alpha * vv.y);
    atomicAdd(&ag[c + 2], alpha * vv.z);
    atomicAdd(&ag[c + 3], alpha * vv.w);
  }
}

// ELU in place (f32) + bf16 recast for next layer's WMMA input
__global__ void elu_bf16_k(float* __restrict__ agg, __bf16* __restrict__ hb, int total) {
  int i = blockIdx.x * blockDim.x + threadIdx.x;
  if (i >= total) return;
  float x = agg[i];
  float y = (x > 0.f) ? x : expm1f(x);
  agg[i] = y;
  hb[i] = (__bf16)y;
}

// ---------------- global attention pooling ----------------

__global__ void pool_init_k(float* __restrict__ gm, float* __restrict__ gs,
                            float* __restrict__ pooled) {
  int i = blockIdx.x * blockDim.x + threadIdx.x;
  if (i < NG) { gm[i] = -INFINITY; gs[i] = 0.f; }
  if (i < NG * 32) pooled[i] = 0.f;
}

__global__ void gate_k(const float* __restrict__ h, const float* __restrict__ Wg,
                       const float* __restrict__ bg, float* __restrict__ gate,
                       float* __restrict__ gm, const int* __restrict__ batch, int n) {
  int nd = (blockIdx.x * blockDim.x + threadIdx.x) >> 5;
  int lane = threadIdx.x & 31;
  if (nd >= n) return;
  float v = h[(size_t)nd * 32 + lane] * Wg[lane];  // fout == 32 == wave width
  for (int off = 16; off; off >>= 1) v += __shfl_down(v, off, 32);
  if (lane == 0) {
    float g = v + bg[0];
    gate[nd] = g;
    atomicMax(&gm[batch[nd]], g);
  }
}

__global__ void gate_exp_k(float* __restrict__ gate, const float* __restrict__ gm,
                           float* __restrict__ gs, const int* __restrict__ batch, int n) {
  int i = blockIdx.x * blockDim.x + threadIdx.x;
  if (i >= n) return;
  int b = batch[i];
  float a = expf(gate[i] - gm[b]);
  gate[i] = a;
  atomicAdd(&gs[b], a);
}

__global__ void pool_scatter_k(const float* __restrict__ h, const float* __restrict__ gate,
                               const float* __restrict__ gs, const int* __restrict__ batch,
                               float* __restrict__ pooled, int n) {
  int nd = (blockIdx.x * blockDim.x + threadIdx.x) >> 5;
  int lane = threadIdx.x & 31;
  if (nd >= n) return;
  int b = batch[nd];
  float coef = gate[nd] / gs[b];
  atomicAdd(&pooled[b * 32 + lane], coef * h[(size_t)nd * 32 + lane]);
}

__global__ void final_out_k(const float* __restrict__ pooled, const float* __restrict__ Wf,
                            const float* __restrict__ bf, float* __restrict__ out) {
  int i = blockIdx.x * blockDim.x + threadIdx.x;
  if (i >= NG * 9) return;
  int g = i / 9, j = i % 9;
  float acc = bf[j];
  for (int c = 0; c < 32; ++c) acc += pooled[g * 32 + c] * Wf[c * 9 + j];
  out[i] = acc;
}

// ---------------- host orchestration ----------------

extern "C" void kernel_launch(void* const* d_in, const int* in_sizes, int n_in,
                              void* d_out, int out_size, void* d_ws, size_t ws_size,
                              hipStream_t stream) {
  (void)in_sizes; (void)n_in; (void)out_size; (void)ws_size;
  const float* x   = (const float*)d_in[0];
  const int* edge  = (const int*)d_in[1];
  const int* src   = edge;            // edge_index[0]
  const int* dst   = edge + NE;       // edge_index[1]
  const int* batch = (const int*)d_in[2];
  const float* Wg  = (const float*)d_in[35];
  const float* bg  = (const float*)d_in[36];
  const float* Wf  = (const float*)d_in[37];
  const float* bfv = (const float*)d_in[38];

  size_t off = 0;
  char* base = (char*)d_ws;
  auto alloc = [&](size_t bytes) -> void* {
    void* p = base + off;
    off += (bytes + 255) & ~(size_t)255;
    return p;
  };
  float*  qkvs   = (float*)alloc((size_t)NN * 2048 * 4);
  __bf16* hb     = (__bf16*)alloc((size_t)NN * 512 * 2);
  float*  agg    = (float*)alloc((size_t)NN * 512 * 4);
  __bf16* wtp    = (__bf16*)alloc((size_t)512 * 2048 * 2);
  float*  bp     = (float*)alloc(2048 * 4);
  float*  logits = (float*)alloc((size_t)NE * 4);   // reused as gate buffer
  float*  ebuf   = (float*)alloc((size_t)NE * 4);
  float*  mbuf   = (float*)alloc(NN * 4);
  float*  sbuf   = (float*)alloc(NN * 4);
  float*  gm     = (float*)alloc(NG * 4);
  float*  gs     = (float*)alloc(NG * 4);
  float*  pooled = (float*)alloc(NG * 32 * 4);

  const int fins[4]  = {128, 512, 256, 64};
  const int fouts[4] = {512, 256, 64, 32};

  // x -> bf16 input
  f32_to_bf16_k<<<(NN * 128 + 255) / 256, 256, 0, stream>>>(x, hb, NN * 128);

  for (int li = 0; li < 4; ++li) {
    int fin = fins[li], fout = fouts[li], Np = 4 * fout;
    const int pb = 3 + li * 8;
    const float* Wq = (const float*)d_in[pb + 0];
    const float* bq = (const float*)d_in[pb + 1];
    const float* Wk = (const float*)d_in[pb + 2];
    const float* bk = (const float*)d_in[pb + 3];
    const float* Wv = (const float*)d_in[pb + 4];
    const float* bv = (const float*)d_in[pb + 5];
    const float* Ws = (const float*)d_in[pb + 6];
    const float* bs = (const float*)d_in[pb + 7];

    int wtot = 4 * fout * fin;
    pack_weights_k<<<(wtot + 255) / 256, 256, 0, stream>>>(
        Wq, Wk, Wv, Ws, bq, bk, bv, bs, wtp, bp, fin, fout);

    // fused Q|K|V|S projection: [10000 x fin] x [fin x 4*fout]
    int m32 = (NN + 31) / 32;                      // 313 32-row tiles
    dim3 ggrid((m32 + 7) / 8, Np / 64);
    wmma_gemm_bias_k<<<ggrid, 256, 0, stream>>>(hb, wtp, bp, qkvs, NN, fin, Np);

    init_node_k<<<(NN + 255) / 256, 256, 0, stream>>>(mbuf, sbuf, NN);

    float scale = 1.0f / sqrtf((float)fout);
    edge_logits_k<<<(NE + 7) / 8, 256, 0, stream>>>(qkvs, src, dst, logits, mbuf,
                                                    NE, fout, Np, scale);
    edge_exp_k<<<(NE + 255) / 256, 256, 0, stream>>>(logits, mbuf, sbuf, ebuf, dst, NE);

    int tot = NN * fout;
    init_agg_skip_k<<<(tot / 4 + 255) / 256, 256, 0, stream>>>(qkvs, agg, fout, Np, tot / 4);
    edge_scatter_k<<<(NE + 7) / 8, 256, 0, stream>>>(qkvs, ebuf, sbuf, src, dst, agg,
                                                     NE, fout, Np);
    elu_bf16_k<<<(tot + 255) / 256, 256, 0, stream>>>(agg, hb, tot);
  }

  // global attention pooling over 16 graphs; h = agg [10000 x 32]
  pool_init_k<<<2, 256, 0, stream>>>(gm, gs, pooled);
  gate_k<<<(NN + 7) / 8, 256, 0, stream>>>(agg, Wg, bg, logits, gm, batch, NN);
  gate_exp_k<<<(NN + 255) / 256, 256, 0, stream>>>(logits, gm, gs, batch, NN);
  pool_scatter_k<<<(NN + 7) / 8, 256, 0, stream>>>(agg, logits, gs, batch, pooled, NN);
  final_out_k<<<1, 256, 0, stream>>>(pooled, Wf, bfv, (float*)d_out);
}
